// MyConv2D_11879879541111
// MI455X (gfx1250) — matrix operand order repaired and checked by reading, not compiled, not run on hardware
//
#include <hip/hip_runtime.h>

// ---------------------------------------------------------------------------
// Conv2D 3x3 SAME, NHWC, fp32 -> implicit GEMM on CDNA5 WMMA (gfx1250).
// M = B*H*W = 100352, N = C_OUT = 256, K = 9*C_IN = 1152.
// fp32 accuracy via bf16 hi/lo split: x*w ~= xh*wh + xh*wl + xl*wh
// using v_wmma_f32_16x16x32_bf16 (wave32).
//
// Weights are pre-split into planar bf16 hi/lo planes laid out [kpos][co][ci]
// in d_ws by a tiny pre-pass kernel; the main kernel streams B tiles into a
// double-buffered LDS stage with the Tensor Data Mover (tensor_load_to_lds,
// TENSORcnt), overlapping the DMA with the WMMA work of the previous chunk.
// ---------------------------------------------------------------------------

#define B_    32
#define H_    56
#define W_    56
#define CI    128
#define CO    256
#define HW_   (H_ * W_)          // 3136
#define MTOT  (B_ * HW_)         // 100352
#define MT    128                // workgroup M tile
#define NT    128                // workgroup N tile
#define KC    32                 // K chunk per LDS stage (one WMMA K)
#define NCHUNK 36                // 9 taps * 4 ci-chunks
#define WELEMS (9 * CO * CI)     // 294912 elements per weight plane

typedef __bf16 bf16_t;
typedef __attribute__((ext_vector_type(16))) __bf16 v16bf;
typedef __attribute__((ext_vector_type(8)))  float  v8f;
typedef __attribute__((ext_vector_type(4)))  unsigned int u32x4;
typedef __attribute__((ext_vector_type(8)))  int          i32x8;
typedef __attribute__((ext_vector_type(4)))  int          i32x4;

union Frag16 {
    uint4 u[2];
    v16bf v;
};

union Half16 {
    bf16_t h[16];
    uint4  u[2];
};

// Load a 16x(bf16) WMMA fragment as two 16-byte LDS reads (ds_load_b128 x2).
__device__ __forceinline__ v16bf ld_frag(const bf16_t* p0, const bf16_t* p1) {
    Frag16 f;
    f.u[0] = *(const uint4*)p0;
    f.u[1] = *(const uint4*)p1;
    return f.v;
}

// LDS byte offset of a __shared__ object (flat addr low 32 bits = LDS offset).
__device__ __forceinline__ unsigned lds_off(const void* p) {
    return (unsigned)(size_t)p;
}

// Issue one 2-D TDM load: tile_dim0 = KC elems (contiguous, 2B each),
// tile_dim1 = NT rows, row stride = rowStride elems. LDS dest is contiguous.
__device__ __forceinline__ void tdm_tile_2d(unsigned ldsOff,
                                            const bf16_t* g,
                                            int rowStride) {
    const unsigned long long ga = (unsigned long long)(size_t)g;
    u32x4 g0;
    g0[0] = 1u;                                              // count=1
    g0[1] = ldsOff;                                          // lds_addr
    g0[2] = (unsigned)ga;                                    // global_addr lo
    g0[3] = ((unsigned)(ga >> 32) & 0x01FFFFFFu) | (2u << 30); // addr hi|type=2
    i32x8 g1;
    g1[0] = 1 << 16;                  // data_size=1 (2 bytes); no flags
    g1[1] = (int)0xFFFF0000u;         // tensor_dim0 lo16 (huge: tile in-bounds)
    g1[2] = (int)0xFFFF0FFFu;         // tensor_dim0 hi16 / tensor_dim1 lo16
    g1[3] = 0x0FFF | (KC << 16);      // tensor_dim1 hi16 / tile_dim0 = 32
    g1[4] = NT;                       // tile_dim1 = 128, tile_dim2 = 0
    g1[5] = rowStride;                // tensor_dim0_stride lo32
    g1[6] = 0;                        // stride hi / dim1_stride lo
    g1[7] = 0;                        // dim1_stride hi
    const i32x4 z4 = {0, 0, 0, 0};
    const i32x8 z8 = {0, 0, 0, 0, 0, 0, 0, 0};
    // 6-arg form (clang-23 / therock-10.0 headers): g0, g1, g2, g3, g4, cpol
    __builtin_amdgcn_tensor_load_to_lds(g0, g1, z4, z4, z8, 0);
}

// ---------------------------------------------------------------------------
// Pre-pass: split fp32 weights into planar bf16 hi/lo, transposed to
// [kpos][co][ci] so main-loop B tiles are contiguous-row 2-D TDM loads.
// ---------------------------------------------------------------------------
__global__ __launch_bounds__(256)
void split_weights(const float* __restrict__ wgt,
                   bf16_t* __restrict__ whi,
                   bf16_t* __restrict__ wlo) {
    const int idx  = blockIdx.x * 256 + threadIdx.x;   // < WELEMS
    const int ci   = idx & (CI - 1);
    const int r    = idx >> 7;
    const int co   = r & (CO - 1);
    const int kpos = r >> 8;
    const float  f = wgt[(kpos * CI + ci) * CO + co];
    const bf16_t h = (bf16_t)f;
    whi[idx] = h;
    wlo[idx] = (bf16_t)(f - (float)h);
}

// ---------------------------------------------------------------------------
// Main kernel.
// ---------------------------------------------------------------------------
__global__ __launch_bounds__(256)
void conv3x3_wmma_bf16x3(const float* __restrict__ x,
                         const bf16_t* __restrict__ whi,
                         const bf16_t* __restrict__ wlo,
                         const float* __restrict__ bias,
                         float* __restrict__ out) {
    // A: single-buffered hi/lo planes [m][ci].
    // B: double-buffered hi/lo planes [co][ci] (ci contiguous) filled by TDM.
    __shared__ bf16_t Ahi[MT][KC];
    __shared__ bf16_t Alo[MT][KC];
    __shared__ bf16_t Bhi[2][NT][KC];
    __shared__ bf16_t Blo[2][NT][KC];

    const int t      = threadIdx.x;      // 0..255
    const int lane   = t & 31;
    const int l15    = lane & 15;
    const int waveId = t >> 5;           // 0..7
    const int wave_m = (waveId & 3) * 32;   // 4 waves along M, 32 rows each
    const int wave_n = (waveId >> 2) * 64;  // 2 waves along N, 64 cols each

    const int mBlock = blockIdx.x * MT;  // 784 blocks, exact
    const int nBlock = blockIdx.y * NT;  // 2 blocks, exact

    // ---- loader-side decode (one output pixel per thread, 2 threads/row) ----
    const int mloc = t & 127;            // 0..127: row of the A tile
    const int half = t >> 7;             // 0/1 : which 16-channel slice
    const int m_g  = mBlock + mloc;
    const int bImg = m_g / HW_;
    const int hw   = m_g - bImg * HW_;
    const int hOut = hw / W_;
    const int wOut = hw - hOut * W_;

    // ---- per-lane fragment addressing (matches ISA bf16 layouts) ----
    const int aK = (lane < 16) ? 0 : 8;   // A lane<16: K{0..7,16..23}; else K{8..15,24..31}
    const int bK = (lane < 16) ? 0 : 8;   // B lane<16: K{0..7,16..23}; else K{8..15,24..31}

    // ---- B-tile source bases in the planar weight planes ----
    // plane layout: [kpos][co][ci]; tile rows = co in [nBlock, nBlock+NT)
    const long  bTileBase = (long)nBlock * CI;        // + kpos*CO*CI + c*KC
    const unsigned ldsBhi0 = lds_off(&Bhi[0][0][0]);
    const unsigned ldsBhi1 = lds_off(&Bhi[1][0][0]);
    const unsigned ldsBlo0 = lds_off(&Blo[0][0][0]);
    const unsigned ldsBlo1 = lds_off(&Blo[1][0][0]);

    // ---- accumulators: 2(M) x 4(N) tiles of 16x16 f32 ----
    v8f acc[2][4];
#pragma unroll
    for (int mt = 0; mt < 2; ++mt)
#pragma unroll
        for (int nt = 0; nt < 4; ++nt)
            acc[mt][nt] = (v8f){0.f, 0.f, 0.f, 0.f, 0.f, 0.f, 0.f, 0.f};

    // ---- prologue: kick off TDM for chunk 0 into buffer 0 ----
    if (waveId == 0) {
        const long g = bTileBase;        // kpos=0, c=0
        tdm_tile_2d(ldsBhi0, whi + g, CI);
        tdm_tile_2d(ldsBlo0, wlo + g, CI);
    }

    // ---- K loop: 36 chunks of 32 (9 taps x 4 ci-chunks) ----
    for (int q = 0; q < NCHUNK; ++q) {
        const int kpos = q >> 2;
        const int c    = q & 3;
        const int buf  = q & 1;

        const int kh  = kpos / 3;
        const int kw  = kpos - kh * 3;
        const int hIn = hOut + kh - 1;
        const int wIn = wOut + kw - 1;
        const bool valid = (hIn >= 0) && (hIn < H_) && (wIn >= 0) && (wIn < W_);

        __syncthreads();   // previous chunk fully consumed (A free, B[buf^1] free)

        // ---- stage A (x) tile: 128 x 32, split into hi/lo bf16 ----
        {
            const float* xp = x + (((long)bImg * H_ + hIn) * W_ + wIn) * CI
                                + c * KC + half * 16;
            Half16 hi, lo;
#pragma unroll
            for (int i = 0; i < 4; ++i) {
                float4 v = valid ? ((const float4*)xp)[i]
                                 : make_float4(0.f, 0.f, 0.f, 0.f);
                const float vv[4] = {v.x, v.y, v.z, v.w};
#pragma unroll
                for (int j = 0; j < 4; ++j) {
                    float  f  = vv[j];
                    bf16_t hb = (bf16_t)f;
                    hi.h[i * 4 + j] = hb;
                    lo.h[i * 4 + j] = (bf16_t)(f - (float)hb);
                }
            }
            uint4* dh = (uint4*)&Ahi[mloc][half * 16];
            uint4* dl = (uint4*)&Alo[mloc][half * 16];
            dh[0] = hi.u[0]; dh[1] = hi.u[1];
            dl[0] = lo.u[0]; dl[1] = lo.u[1];
        }

        // ---- wave0: make sure the in-flight B tile for this chunk landed ----
        if (waveId == 0) {
            __builtin_amdgcn_s_wait_tensorcnt(0);
        }

        __syncthreads();   // A staged + B[buf] ready for everyone

        // ---- wave0: launch TDM for the next chunk into the other buffer ----
        if (waveId == 0 && (q + 1) < NCHUNK) {
            const int qn    = q + 1;
            const int kposN = qn >> 2;
            const int cN    = qn & 3;
            const long g    = (long)kposN * CO * CI + bTileBase + (long)cN * KC;
            tdm_tile_2d((qn & 1) ? ldsBhi1 : ldsBhi0, whi + g, CI);
            tdm_tile_2d((qn & 1) ? ldsBlo1 : ldsBlo0, wlo + g, CI);
        }

        // ---- load fragments (two ds_load_b128 each) ----
        v16bf afh[2], afl[2];
#pragma unroll
        for (int mt = 0; mt < 2; ++mt) {
            const int row = wave_m + mt * 16 + l15;
            afh[mt] = ld_frag(&Ahi[row][aK], &Ahi[row][aK + 16]);
            afl[mt] = ld_frag(&Alo[row][aK], &Alo[row][aK + 16]);
        }
        v16bf bfh[4], bfl[4];
#pragma unroll
        for (int nt = 0; nt < 4; ++nt) {
            const int col = wave_n + nt * 16 + l15;
            bfh[nt] = ld_frag(&Bhi[buf][col][bK], &Bhi[buf][col][bK + 16]);
            bfl[nt] = ld_frag(&Blo[buf][col][bK], &Blo[buf][col][bK + 16]);
        }

        // ---- 8 tiles x 3 split terms = 24 WMMAs per wave per chunk ----
#pragma unroll
        for (int mt = 0; mt < 2; ++mt) {
#pragma unroll
            for (int nt = 0; nt < 4; ++nt) {
                acc[mt][nt] = __builtin_amdgcn_wmma_f32_16x16x32_bf16(
                    false, afh[mt], false, bfh[nt],
                    (short)0, acc[mt][nt], false, false);
                acc[mt][nt] = __builtin_amdgcn_wmma_f32_16x16x32_bf16(
                    false, afh[mt], false, bfl[nt],
                    (short)0, acc[mt][nt], false, false);
                acc[mt][nt] = __builtin_amdgcn_wmma_f32_16x16x32_bf16(
                    false, afl[mt], false, bfh[nt],
                    (short)0, acc[mt][nt], false, false);
            }
        }
    }

    // ---- epilogue: bias + store per C-matrix VGPR layout ----
    // acc VGPR r: lanes 0-15 -> row (tile+r), lanes 16-31 -> row (tile+8+r),
    // column = tile_n + (lane & 15).
    const int rOff = (lane < 16) ? 0 : 8;
#pragma unroll
    for (int nt = 0; nt < 4; ++nt) {
        const int nG = nBlock + wave_n + nt * 16 + l15;
        const float bv = bias[nG];
#pragma unroll
        for (int mt = 0; mt < 2; ++mt) {
            const int mBase = mBlock + wave_m + mt * 16 + rOff;
            v8f a = acc[mt][nt];
#pragma unroll
            for (int r = 0; r < 8; ++r) {
                out[(long)(mBase + r) * CO + nG] = a[r] + bv;
            }
        }
    }
}

extern "C" void kernel_launch(void* const* d_in, const int* in_sizes, int n_in,
                              void* d_out, int out_size, void* d_ws, size_t ws_size,
                              hipStream_t stream) {
    const float* x    = (const float*)d_in[0];
    const float* wgt  = (const float*)d_in[1];
    const float* bias = (const float*)d_in[2];
    float* out        = (float*)d_out;

    bf16_t* whi = (bf16_t*)d_ws;              // WELEMS bf16
    bf16_t* wlo = whi + WELEMS;               // WELEMS bf16 (total ~1.18 MB)

    split_weights<<<dim3(WELEMS / 256), dim3(256), 0, stream>>>(wgt, whi, wlo);

    dim3 grid(MTOT / MT, CO / NT);   // 784 x 2
    dim3 block(256);                 // 8 wave32s
    conv3x3_wmma_bf16x3<<<grid, block, 0, stream>>>(x, whi, wlo, bias, out);
}
